// E3Conv_27754078667284
// MI455X (gfx1250) — compile-verified, matrix operand order checked by reading, not run on hardware
//
#include <hip/hip_runtime.h>

typedef __attribute__((ext_vector_type(16))) _Float16 v16h;
typedef __attribute__((ext_vector_type(8)))  float    v8f;

#define N_NODES_C 10000
#define N_EDGES_C 160000
#define COLS 1536      // 3*8*8*8
#define NTILES 96      // COLS/16
#define FEAT 72        // 8*1 + 8*3 + 8*5

// SiLU with hardware reciprocal (v_rcp_f32) instead of exact IEEE division:
// avoids the v_div_scale/v_div_fmas/v_div_fixup chains in the WMMA epilogues.
__device__ __forceinline__ float silu_f(float x) {
    return x * __builtin_amdgcn_rcpf(1.0f + __expf(-x));
}

// A-fragment K-index map for v_wmma_f32_16x16x32_f16 (16-bit A, 16x32):
// lane L: row M=L%16, K-base = (L<16?0:8); VGPR pair p -> K = (p<4 ? 2p : 16+2(p-4)) + base
__device__ __forceinline__ int frag_k0(int p, int base) {
    return (p < 4 ? 2 * p : 16 + 2 * (p - 4)) + base;
}

__device__ __forceinline__ v16h pack_frag(const float* row, int base, int koff) {
    v16h a;
    #pragma unroll
    for (int p = 0; p < 8; ++p) {
        int k0 = frag_k0(p, base) + koff;
        a[2 * p]     = (_Float16)row[k0];
        a[2 * p + 1] = (_Float16)row[k0 + 1];
    }
    return a;
}

// layer-0 input is 8-wide; pad K to 32 with zeros
__device__ __forceinline__ v16h pack_frag_pad8(const float* row, int base) {
    v16h a;
    #pragma unroll
    for (int p = 0; p < 8; ++p) {
        int k0 = frag_k0(p, base);
        a[2 * p]     = (k0     < 8) ? (_Float16)row[k0]     : (_Float16)0.0f;
        a[2 * p + 1] = (k0 + 1 < 8) ? (_Float16)row[k0 + 1] : (_Float16)0.0f;
    }
    return a;
}

// ---------------------------------------------------------------- zero
__global__ void zero_f(float* p, int n) {
    int i = blockIdx.x * blockDim.x + threadIdx.x;
    if (i < n) p[i] = 0.0f;
}

// ---------------------------------------------------------------- K0: node MLP -> Ai (N_NODES x 8)
__global__ void node_mlp(const int* __restrict__ A, const float* __restrict__ emb,
                         const float* __restrict__ W0, const float* __restrict__ b0,
                         const float* __restrict__ W1, const float* __restrict__ b1,
                         const float* __restrict__ W2, const float* __restrict__ b2,
                         float* __restrict__ Ai) {
    int i = blockIdx.x * blockDim.x + threadIdx.x;
    if (i >= N_NODES_C) return;
    int a = A[i];
    float e[16];
    #pragma unroll
    for (int j = 0; j < 16; ++j) e[j] = emb[a * 16 + j];
    float h1[64];
    for (int o = 0; o < 64; ++o) {
        float acc = b0[o];
        #pragma unroll
        for (int j = 0; j < 16; ++j) acc += e[j] * W0[j * 64 + o];
        h1[o] = silu_f(acc);
    }
    float h2[32];
    for (int o = 0; o < 32; ++o) {
        float acc = b1[o];
        for (int j = 0; j < 64; ++j) acc += h1[j] * W1[j * 32 + o];
        h2[o] = silu_f(acc);
    }
    #pragma unroll
    for (int o = 0; o < 8; ++o) {
        float acc = b2[o];
        for (int j = 0; j < 32; ++j) acc += h2[j] * W2[j * 8 + o];
        Ai[i * 8 + o] = acc;
    }
}

// ---------------------------------------------------------------- K1: generic f32->f16 B-fragment packer
// dst[((t*nki + ki)*32 + lane)*16 + h] ; W is (K x fan_out) row-major with row stride `stride`;
// k >= kmax reads as 0 (zero padding for layer-0).
__global__ void pack_w(const float* __restrict__ W, _Float16* __restrict__ dst,
                       int ncolt, int nki, int stride, int kmax) {
    int idx = blockIdx.x * blockDim.x + threadIdx.x;
    if (idx >= ncolt * nki * 32 * 16) return;
    int h    = idx & 15;
    int lane = (idx >> 4) & 31;
    int ki   = (idx >> 9) % nki;
    int t    = (idx >> 9) / nki;
    int p    = h >> 1;
    int klocal = (p < 4 ? 2 * p : 16 + 2 * (p - 4)) + ((lane < 16) ? 0 : 8) + (h & 1);
    int k = ki * 32 + klocal;
    int c = t * 16 + (lane & 15);
    dst[idx] = (k < kmax) ? (_Float16)W[k * stride + c] : (_Float16)0.0f;
}

// ---------------------------------------------------------------- K2: fused edge kernel. One wave per 16 edges.
__global__ void __launch_bounds__(32)
edge_kernel(const float* __restrict__ pos, const int* __restrict__ batch,
            const int* __restrict__ esrc, const int* __restrict__ edst,
            const float* __restrict__ shifts, const float* __restrict__ cell,
            const float* __restrict__ fcb0, const float* __restrict__ fcb1,
            const float* __restrict__ fcb2, const float* __restrict__ fcb3,
            const float* __restrict__ Ai,
            const _Float16* __restrict__ Bp0, const _Float16* __restrict__ Bp1,
            const _Float16* __restrict__ Bp2, const _Float16* __restrict__ Bp3,
            float* __restrict__ out, float* __restrict__ counts) {
    __shared__ float sh_s[16][9];
    __shared__ float asrcT[8][16];   // transposed: [u][edge]
    __shared__ float sdstT[8][16];   // transposed: [v][edge]
    __shared__ float rad_s[16][8];
    __shared__ float hA[16][64];
    __shared__ float hB[16][64];
    __shared__ int   dst_s[16];

    const int L     = threadIdx.x;
    const int e0    = blockIdx.x * 16;
    const int N     = L & 15;
    const int half  = L >> 4;
    const int ebase = half * 8;      // D rows this lane holds: ebase..ebase+7
    const int buf   = N >> 3;        // v parity
    const int wo    = N & 7;

    // ---------------- geometry / radial / per-edge vectors (lanes 0..15)
    if (L < 16) {
        int e = e0 + L;
        int src = esrc[e], dst = edst[e];
        dst_s[L] = dst;
        int g = batch[src];
        const float* C = cell + g * 9;
        float s0 = shifts[e * 3 + 0], s1 = shifts[e * 3 + 1], s2 = shifts[e * 3 + 2];
        float vx = pos[dst * 3 + 0] - pos[src * 3 + 0] + s0 * C[0] + s1 * C[3] + s2 * C[6];
        float vy = pos[dst * 3 + 1] - pos[src * 3 + 1] + s0 * C[1] + s1 * C[4] + s2 * C[7];
        float vz = pos[dst * 3 + 2] - pos[src * 3 + 2] + s0 * C[2] + s1 * C[5] + s2 * C[8];
        float r = sqrtf(vx * vx + vy * vy + vz * vz);
        float inv = 1.0f / fmaxf(r, 1e-12f);
        float x = vx * inv, y = vy * inv, z = vz * inv;
        const float c3 = 1.7320508075688772f, c15 = 3.872983346207417f, c5 = 2.23606797749979f;
        sh_s[L][0] = 1.0f;
        sh_s[L][1] = c3 * x;  sh_s[L][2] = c3 * y;  sh_s[L][3] = c3 * z;
        sh_s[L][4] = c15 * x * z;  sh_s[L][5] = c15 * x * y;
        sh_s[L][6] = c5 * (y * y - 0.5f * (x * x + z * z));
        sh_s[L][7] = c15 * y * z;  sh_s[L][8] = 0.5f * c15 * (z * z - x * x);
        float xr = fminf(fmaxf(r * 0.125f, 0.0f), 1.0f);
        float gate = (r <= 8.0f) ? 2.8284271247461903f : 0.0f;   // sqrt(8) * (r<=R_MAX)
        #pragma unroll
        for (int i = 0; i < 8; ++i) {
            float d = (xr - (float)i * (1.0f / 7.0f)) * 7.0f;
            rad_s[L][i] = __expf(-0.5f * d * d) * gate;
        }
        #pragma unroll
        for (int i = 0; i < 8; ++i) {
            asrcT[i][L] = Ai[src * 8 + i];
            sdstT[i][L] = Ai[dst * 8 + i];
        }
        atomicAdd(&counts[dst], 1.0f);
    }
    __syncthreads();

    // ---------------- g-MLP on WMMA: 16 edges x (8->64->64->64), SiLU between
    const int base = half ? 8 : 0;
    {   // layer 0 : K=32 (zero-padded from 8)
        v16h a = pack_frag_pad8(&rad_s[N][0], base);
        #pragma unroll
        for (int ct = 0; ct < 4; ++ct) {
            v16h b = *(const v16h*)(Bp0 + ((size_t)ct * 32 + L) * 16);
            v8f c = {};
            c = __builtin_amdgcn_wmma_f32_16x16x32_f16(false, a, false, b, (short)0, c, false, false);
            int o = ct * 16 + N;
            float bias = fcb0[o];
            #pragma unroll
            for (int r = 0; r < 8; ++r) hA[ebase + r][o] = silu_f(c[r] + bias);
        }
    }
    __syncthreads();
    {   // layer 1 : K=64
        v16h a0 = pack_frag(&hA[N][0], base, 0);
        v16h a1 = pack_frag(&hA[N][0], base, 32);
        #pragma unroll
        for (int ct = 0; ct < 4; ++ct) {
            v16h b0 = *(const v16h*)(Bp1 + ((size_t)(ct * 2 + 0) * 32 + L) * 16);
            v16h b1 = *(const v16h*)(Bp1 + ((size_t)(ct * 2 + 1) * 32 + L) * 16);
            v8f c = {};
            c = __builtin_amdgcn_wmma_f32_16x16x32_f16(false, a0, false, b0, (short)0, c, false, false);
            c = __builtin_amdgcn_wmma_f32_16x16x32_f16(false, a1, false, b1, (short)0, c, false, false);
            int o = ct * 16 + N;
            float bias = fcb1[o];
            #pragma unroll
            for (int r = 0; r < 8; ++r) hB[ebase + r][o] = silu_f(c[r] + bias);
        }
    }
    __syncthreads();
    {   // layer 2 : K=64
        v16h a0 = pack_frag(&hB[N][0], base, 0);
        v16h a1 = pack_frag(&hB[N][0], base, 32);
        #pragma unroll
        for (int ct = 0; ct < 4; ++ct) {
            v16h b0 = *(const v16h*)(Bp2 + ((size_t)(ct * 2 + 0) * 32 + L) * 16);
            v16h b1 = *(const v16h*)(Bp2 + ((size_t)(ct * 2 + 1) * 32 + L) * 16);
            v8f c = {};
            c = __builtin_amdgcn_wmma_f32_16x16x32_f16(false, a0, false, b0, (short)0, c, false, false);
            c = __builtin_amdgcn_wmma_f32_16x16x32_f16(false, a1, false, b1, (short)0, c, false, false);
            int o = ct * 16 + N;
            float bias = fcb2[o];
            #pragma unroll
            for (int r = 0; r < 8; ++r) hA[ebase + r][o] = silu_f(c[r] + bias);   // g in hA
        }
    }
    __syncthreads();

    // ---------------- A fragments for g (16x64)
    v16h a0 = pack_frag(&hA[N][0], base, 0);
    v16h a1 = pack_frag(&hA[N][0], base, 32);

    // ---------------- main loop: w = g @ W3 + b3 tile-by-tile, fully linear contraction
    // R[l][r] accumulates sum_{u,v} (w + bias) * s[v] * Asrc[u]  for this lane's v-parity
    float R[3][8];
    #pragma unroll
    for (int l = 0; l < 3; ++l)
        #pragma unroll
        for (int r = 0; r < 8; ++r) R[l][r] = 0.0f;

    for (int lu = 0; lu < 24; ++lu) {        // lu = l*8 + u
        const int l = lu >> 3, u = lu & 7;
        float4 au0 = *(const float4*)&asrcT[u][ebase];
        float4 au1 = *(const float4*)&asrcT[u][ebase + 4];
        #pragma unroll
        for (int tv = 0; tv < 4; ++tv) {
            int t = lu * 4 + tv;
            int v = 2 * tv + buf;            // this lane's v for this tile
            v16h b0 = *(const v16h*)(Bp3 + ((size_t)(t * 2 + 0) * 32 + L) * 16);
            v16h b1 = *(const v16h*)(Bp3 + ((size_t)(t * 2 + 1) * 32 + L) * 16);
            v8f c = {};
            c = __builtin_amdgcn_wmma_f32_16x16x32_f16(false, a0, false, b0, (short)0, c, false, false);
            c = __builtin_amdgcn_wmma_f32_16x16x32_f16(false, a1, false, b1, (short)0, c, false, false);
            float bias = fcb3[t * 16 + N];
            float4 sv0 = *(const float4*)&sdstT[v][ebase];
            float4 sv1 = *(const float4*)&sdstT[v][ebase + 4];
            float f[8] = { sv0.x * au0.x, sv0.y * au0.y, sv0.z * au0.z, sv0.w * au0.w,
                           sv1.x * au1.x, sv1.y * au1.y, sv1.z * au1.z, sv1.w * au1.w };
            #pragma unroll
            for (int r = 0; r < 8; ++r) R[l][r] += (c[r] + bias) * f[r];
        }
    }

    // ---------------- combine v-parity halves (lanes N and N^8 share (e-half, wo))
    #pragma unroll
    for (int l = 0; l < 3; ++l)
        #pragma unroll
        for (int r = 0; r < 8; ++r) R[l][r] += __shfl_xor(R[l][r], 8, 32);

    // ---------------- scatter-mean accumulate (one parity lane per (e-half, wo))
    if (buf == 0) {
        const float norm = 0.125f;   // 1/sqrt(MUL*MUL)
        const int foff[3] = {0, 8, 32}, mcnt[3] = {1, 3, 5}, aoff[3] = {0, 1, 4};
        #pragma unroll
        for (int r = 0; r < 8; ++r) {
            int e = ebase + r;
            size_t obase = (size_t)dst_s[e] * FEAT;
            #pragma unroll
            for (int l = 0; l < 3; ++l) {
                float Rn = R[l][r] * norm;
                for (int m = 0; m < mcnt[l]; ++m)
                    atomicAdd(&out[obase + foff[l] + wo * mcnt[l] + m], Rn * sh_s[e][aoff[l] + m]);
            }
        }
    }
}

// ---------------------------------------------------------------- K3: segment mean
__global__ void finalize_k(float* __restrict__ out, const float* __restrict__ counts) {
    int i = blockIdx.x * blockDim.x + threadIdx.x;
    if (i >= N_NODES_C * FEAT) return;
    float c = fmaxf(counts[i / FEAT], 1.0f);
    out[i] = out[i] / c;
}

// ---------------------------------------------------------------- launch
extern "C" void kernel_launch(void* const* d_in, const int* in_sizes, int n_in,
                              void* d_out, int out_size, void* d_ws, size_t ws_size,
                              hipStream_t stream) {
    const float* pos    = (const float*)d_in[0];
    const int*   A      = (const int*)d_in[1];
    const int*   batch  = (const int*)d_in[2];
    const int*   esrc   = (const int*)d_in[3];
    const int*   edst   = (const int*)d_in[4];
    const float* shifts = (const float*)d_in[5];
    const float* cell   = (const float*)d_in[6];
    const float* emb    = (const float*)d_in[7];
    const float* fitW0 = (const float*)d_in[8];  const float* fitb0 = (const float*)d_in[9];
    const float* fitW1 = (const float*)d_in[10]; const float* fitb1 = (const float*)d_in[11];
    const float* fitW2 = (const float*)d_in[12]; const float* fitb2 = (const float*)d_in[13];
    const float* fcW0  = (const float*)d_in[14]; const float* fcb0  = (const float*)d_in[15];
    const float* fcW1  = (const float*)d_in[16]; const float* fcb1  = (const float*)d_in[17];
    const float* fcW2  = (const float*)d_in[18]; const float* fcb2  = (const float*)d_in[19];
    const float* fcW3  = (const float*)d_in[20]; const float* fcb3  = (const float*)d_in[21];

    char* ws = (char*)d_ws;
    float*    Ai_ws  = (float*)ws;                 // 10000*8*4          = 320000 B
    _Float16* Bp3    = (_Float16*)(ws + 320000);   // 96*2*512*2         = 196608 B
    _Float16* Bp1    = (_Float16*)(ws + 516608);   // 4*2*512*2          =   8192 B
    _Float16* Bp2    = (_Float16*)(ws + 524800);   // 4*2*512*2          =   8192 B
    _Float16* Bp0    = (_Float16*)(ws + 532992);   // 4*1*512*2          =   4096 B
    float*    cnt_ws = (float*)(ws + 537088);      // 10000*4            =  40000 B
    float*    out    = (float*)d_out;              // 10000*72 f32

    zero_f<<<(N_NODES_C * FEAT + 255) / 256, 256, 0, stream>>>(out, N_NODES_C * FEAT);
    zero_f<<<(N_NODES_C + 255) / 256, 256, 0, stream>>>(cnt_ws, N_NODES_C);
    node_mlp<<<(N_NODES_C + 127) / 128, 128, 0, stream>>>(A, emb, fitW0, fitb0, fitW1, fitb1,
                                                          fitW2, fitb2, Ai_ws);
    pack_w<<<(NTILES * 2 * 512 + 255) / 256, 256, 0, stream>>>(fcW3, Bp3, NTILES, 2, COLS, 64);
    pack_w<<<(4 * 2 * 512 + 255) / 256, 256, 0, stream>>>(fcW1, Bp1, 4, 2, 64, 64);
    pack_w<<<(4 * 2 * 512 + 255) / 256, 256, 0, stream>>>(fcW2, Bp2, 4, 2, 64, 64);
    pack_w<<<(4 * 1 * 512 + 255) / 256, 256, 0, stream>>>(fcW0, Bp0, 4, 1, 64, 8);
    edge_kernel<<<N_EDGES_C / 16, 32, 0, stream>>>(pos, batch, esrc, edst, shifts, cell,
                                                   fcb0, fcb1, fcb2, fcb3,
                                                   Ai_ws, Bp0, Bp1, Bp2, Bp3, out, cnt_ws);
    finalize_k<<<(N_NODES_C * FEAT + 255) / 256, 256, 0, stream>>>(out, cnt_ws);
}
